// QuantMatMul_72112500899891
// MI455X (gfx1250) — 8-bit matrix operand order repaired and checked by reading, compiled, not run on hardware
//
#include <hip/hip_runtime.h>

typedef int v8i __attribute__((ext_vector_type(8)));

#define K_DIM 4096
#define N_DIM 4096
#define M_DIM 16384

__device__ __forceinline__ int q8(float x, float inv, float zp) {
    float f = rintf(x * inv) + zp;
    int q = (int)f;
    return q < 0 ? 0 : (q > 255 ? 255 : q);
}

__device__ __forceinline__ int sumbytes(unsigned v) {
    return (int)((v & 255u) + ((v >> 8) & 255u) + ((v >> 16) & 255u) + (v >> 24));
}

__device__ __forceinline__ unsigned to_lds(const void* p) {
    return (unsigned)(size_t)p;
}

__device__ __forceinline__ void async_ld16(unsigned lds_off, const unsigned char* gptr) {
    unsigned long long ga = (unsigned long long)(size_t)gptr;
    asm volatile("global_load_async_to_lds_b128 %0, %1, off"
                 :: "v"(lds_off), "v"(ga)
                 : "memory");
}

__global__ __launch_bounds__(256) void zero_i32(int* p, int n) {
    for (int i = blockIdx.x * 256 + threadIdx.x; i < n; i += gridDim.x * 256) p[i] = 0;
}

__global__ __launch_bounds__(256) void minmax_partial(const float* __restrict__ X, int n4,
                                                      float2* __restrict__ part) {
    __shared__ float smn[256], smx[256];
    const float4* X4 = (const float4*)X;
    int tid = threadIdx.x;
    float mn = 3.402823466e38f, mx = -3.402823466e38f;
    for (int i = blockIdx.x * 256 + tid; i < n4; i += gridDim.x * 256) {
        float4 v = X4[i];
        mn = fminf(mn, fminf(fminf(v.x, v.y), fminf(v.z, v.w)));
        mx = fmaxf(mx, fmaxf(fmaxf(v.x, v.y), fmaxf(v.z, v.w)));
    }
    smn[tid] = mn; smx[tid] = mx;
    __syncthreads();
    for (int s = 128; s > 0; s >>= 1) {
        if (tid < s) { smn[tid] = fminf(smn[tid], smn[tid + s]); smx[tid] = fmaxf(smx[tid], smx[tid + s]); }
        __syncthreads();
    }
    if (tid == 0) part[blockIdx.x] = make_float2(smn[0], smx[0]);
}

__global__ __launch_bounds__(256) void finalize_qp(const float2* __restrict__ pA, int nA,
                                                   const float2* __restrict__ pB, int nB,
                                                   float* __restrict__ qp) {
    __shared__ float smn[256], smx[256];
    int tid = threadIdx.x;
    for (int which = 0; which < 2; ++which) {
        const float2* p = which ? pB : pA;
        int n = which ? nB : nA;
        float mn = 3.402823466e38f, mx = -3.402823466e38f;
        for (int i = tid; i < n; i += 256) { mn = fminf(mn, p[i].x); mx = fmaxf(mx, p[i].y); }
        smn[tid] = mn; smx[tid] = mx;
        __syncthreads();
        for (int s = 128; s > 0; s >>= 1) {
            if (tid < s) { smn[tid] = fminf(smn[tid], smn[tid + s]); smx[tid] = fmaxf(smx[tid], smx[tid + s]); }
            __syncthreads();
        }
        if (tid == 0) {
            float scale = fmaxf((smx[0] - smn[0]) / 255.0f, 1e-8f);
            float zp = rintf(-smn[0] / scale);
            qp[which * 2 + 0] = scale;
            qp[which * 2 + 1] = zp;
        }
        __syncthreads();
    }
}

__global__ __launch_bounds__(256) void quant_rows(const float* __restrict__ X,
                                                  const float* __restrict__ qp,
                                                  unsigned char* __restrict__ Xq,
                                                  int* __restrict__ rowsum) {
    int wave = threadIdx.x >> 5, lane = threadIdx.x & 31;
    int row = blockIdx.x * 8 + wave;
    float s = qp[0], zp = qp[1];
    float inv = 1.0f / s;
    const float4* X4 = (const float4*)(X + (size_t)row * K_DIM);
    unsigned* Q = (unsigned*)(Xq + (size_t)row * K_DIM);
    int sum = 0;
#pragma unroll 4
    for (int j = 0; j < 32; ++j) {
        float4 v = X4[j * 32 + lane];
        int q0 = q8(v.x, inv, zp), q1 = q8(v.y, inv, zp);
        int q2 = q8(v.z, inv, zp), q3 = q8(v.w, inv, zp);
        Q[j * 32 + lane] = (unsigned)q0 | ((unsigned)q1 << 8) | ((unsigned)q2 << 16) | ((unsigned)q3 << 24);
        sum += q0 + q1 + q2 + q3;
    }
#pragma unroll
    for (int m = 16; m > 0; m >>= 1) sum += __shfl_xor(sum, m, 32);
    if (lane == 0) rowsum[row] = sum;
}

__global__ __launch_bounds__(256) void quantB_t(const float* __restrict__ B,
                                                const float* __restrict__ qp,
                                                unsigned char* __restrict__ Bqt,
                                                int* __restrict__ colB) {
    __shared__ __align__(16) unsigned char t[64 * 80];
    __shared__ int csum[64];
    int tid = threadIdx.x;
    int nBase = blockIdx.x * 64, kBase = blockIdx.y * 64;
    float s = qp[2], zp = qp[3];
    float inv = 1.0f / s;
    if (tid < 64) csum[tid] = 0;

    int kl = tid >> 2;
    int nc = (tid & 3) << 4;
    const float4* Brow = (const float4*)(B + (size_t)(kBase + kl) * N_DIM + nBase + nc);
#pragma unroll
    for (int j = 0; j < 4; ++j) {
        float4 v = Brow[j];
        t[(nc + j * 4 + 0) * 80 + kl] = (unsigned char)q8(v.x, inv, zp);
        t[(nc + j * 4 + 1) * 80 + kl] = (unsigned char)q8(v.y, inv, zp);
        t[(nc + j * 4 + 2) * 80 + kl] = (unsigned char)q8(v.z, inv, zp);
        t[(nc + j * 4 + 3) * 80 + kl] = (unsigned char)q8(v.w, inv, zp);
    }
    __syncthreads();

    int r = tid >> 2;
    int c = (tid & 3) << 4;
    uint4 u = *(const uint4*)(&t[r * 80 + c]);
    *(uint4*)(&Bqt[(size_t)(nBase + r) * K_DIM + kBase + c]) = u;
    int ssum = sumbytes(u.x) + sumbytes(u.y) + sumbytes(u.z) + sumbytes(u.w);
    atomicAdd(&csum[r], ssum);
    __syncthreads();
    if (tid < 64) atomicAdd(&colB[nBase + tid], csum[tid]);
}

__global__ __launch_bounds__(256) void wmma_gemm(const unsigned char* __restrict__ Aq,
                                                 const unsigned char* __restrict__ Bqt,
                                                 const int* __restrict__ rowA,
                                                 const int* __restrict__ colB,
                                                 const float* __restrict__ qp,
                                                 float* __restrict__ out) {
    __shared__ __align__(16) unsigned char At[2][128 * 64];
    __shared__ __align__(16) unsigned char Bt[2][128 * 64];

    int tid = threadIdx.x;
    int mBase = blockIdx.y * 128;
    int nBase = blockIdx.x * 128;
    int wave = tid >> 5, lane = tid & 31;
    int mw = wave & 3;
    int nw = wave >> 2;
    int m_l = lane & 15, hi = lane >> 4;

    int sidx0 = tid, sidx1 = 256 + tid;
    int sr0 = sidx0 >> 2, sc0 = (sidx0 & 3) << 4;
    int sr1 = sidx1 >> 2, sc1 = (sidx1 & 3) << 4;
    unsigned ldsA = to_lds(&At[0][0]);
    unsigned ldsB = to_lds(&Bt[0][0]);

    v8i acc[2][4];
    v8i vzero = {0, 0, 0, 0, 0, 0, 0, 0};
#pragma unroll
    for (int mf = 0; mf < 2; ++mf)
#pragma unroll
        for (int nf = 0; nf < 4; ++nf) acc[mf][nf] = vzero;

    const int NT = K_DIM / 64;

    {
        int kt = 0;
        async_ld16(ldsA + sr0 * 64 + sc0, Aq + (size_t)(mBase + sr0) * K_DIM + kt + sc0);
        async_ld16(ldsB + sr0 * 64 + sc0, Bqt + (size_t)(nBase + sr0) * K_DIM + kt + sc0);
        async_ld16(ldsA + sr1 * 64 + sc1, Aq + (size_t)(mBase + sr1) * K_DIM + kt + sc1);
        async_ld16(ldsB + sr1 * 64 + sc1, Bqt + (size_t)(nBase + sr1) * K_DIM + kt + sc1);
    }

    for (int t = 0; t < NT; ++t) {
        int buf = t & 1;
        if (t + 1 < NT) {
            int kt = (t + 1) * 64;
            unsigned lo = (buf ^ 1) * (unsigned)(128 * 64);
            async_ld16(ldsA + lo + sr0 * 64 + sc0, Aq + (size_t)(mBase + sr0) * K_DIM + kt + sc0);
            async_ld16(ldsB + lo + sr0 * 64 + sc0, Bqt + (size_t)(nBase + sr0) * K_DIM + kt + sc0);
            async_ld16(ldsA + lo + sr1 * 64 + sc1, Aq + (size_t)(mBase + sr1) * K_DIM + kt + sc1);
            async_ld16(ldsB + lo + sr1 * 64 + sc1, Bqt + (size_t)(nBase + sr1) * K_DIM + kt + sc1);
            asm volatile("s_wait_asynccnt 0x4" ::: "memory");
        } else {
            asm volatile("s_wait_asynccnt 0x0" ::: "memory");
        }
        __syncthreads();

        v8i a[2];
#pragma unroll
        for (int mf = 0; mf < 2; ++mf) {
            int row = mw * 32 + mf * 16 + m_l;
#pragma unroll
            for (int p = 0; p < 4; ++p) {
                uint2 u = *(const uint2*)(&At[buf][row * 64 + (p >> 1) * 32 + hi * 16 + (p & 1) * 8]);
                a[mf][2 * p + 0] = (int)u.x;
                a[mf][2 * p + 1] = (int)u.y;
            }
        }
        v8i b[4];
#pragma unroll
        for (int nf = 0; nf < 4; ++nf) {
            int nrow = nw * 64 + nf * 16 + m_l;
#pragma unroll
            for (int q = 0; q < 2; ++q) {
                uint4 u = *(const uint4*)(&Bt[buf][nrow * 64 + q * 32 + hi * 16]);
                b[nf][4 * q + 0] = (int)u.x;
                b[nf][4 * q + 1] = (int)u.y;
                b[nf][4 * q + 2] = (int)u.z;
                b[nf][4 * q + 3] = (int)u.w;
            }
        }

#pragma unroll
        for (int mf = 0; mf < 2; ++mf)
#pragma unroll
            for (int nf = 0; nf < 4; ++nf)
                acc[mf][nf] = __builtin_amdgcn_wmma_i32_16x16x64_iu8(
                    false, a[mf], false, b[nf], acc[mf][nf], false, false);

        __syncthreads();
    }

    float sA = qp[0], zA = qp[1], sB = qp[2], zB = qp[3];
    int izA = (int)zA, izB = (int)zB;
    float sc = sA * sB;
    int corr = K_DIM * izA * izB;
#pragma unroll
    for (int mf = 0; mf < 2; ++mf) {
#pragma unroll
        for (int nf = 0; nf < 4; ++nf) {
            int n = nBase + nw * 64 + nf * 16 + m_l;
            int cB = colB[n];
#pragma unroll
            for (int r = 0; r < 8; ++r) {
                int m = mBase + mw * 32 + mf * 16 + hi * 8 + r;
                int v = acc[mf][nf][r] - izB * rowA[m] - izA * cB + corr;
                out[(size_t)m * N_DIM + n] = sc * (float)v;
            }
        }
    }
}

extern "C" void kernel_launch(void* const* d_in, const int* in_sizes, int n_in,
                              void* d_out, int out_size, void* d_ws, size_t ws_size,
                              hipStream_t stream) {
    const float* A = (const float*)d_in[0];
    const float* B = (const float*)d_in[1];
    float* out = (float*)d_out;

    char* ws = (char*)d_ws;
    float*  qp   = (float*)(ws + 0);
    float2* pA   = (float2*)(ws + 1024);
    float2* pB   = (float2*)(ws + 1024 + 4096);
    int*    rowA = (int*)(ws + 8192);
    int*    colB = (int*)(ws + 8192 + 65536);
    unsigned char* Aq  = (unsigned char*)(ws + 131072);
    unsigned char* Bqt = (unsigned char*)(ws + 131072 + (size_t)64 * 1024 * 1024);

    zero_i32<<<16, 256, 0, stream>>>(colB, N_DIM);
    minmax_partial<<<512, 256, 0, stream>>>(A, (M_DIM * K_DIM) / 4, pA);
    minmax_partial<<<128, 256, 0, stream>>>(B, (K_DIM * N_DIM) / 4, pB);
    finalize_qp<<<1, 256, 0, stream>>>(pA, 512, pB, 128, qp);
    quant_rows<<<M_DIM / 8, 256, 0, stream>>>(A, qp, Aq, rowA);
    quantB_t<<<dim3(N_DIM / 64, K_DIM / 64), 256, 0, stream>>>(B, qp, Bqt, colB);
    wmma_gemm<<<dim3(N_DIM / 128, M_DIM / 128), 256, 0, stream>>>(Aq, Bqt, rowA, colB, qp, out);
}
